// NewSupConLoss_26250840113505
// MI455X (gfx1250) — compile-verified, matrix operand order checked by reading, not run on hardware
//
#include <hip/hip_runtime.h>
#include <hip/hip_bf16.h>
#include <stdint.h>

// Problem constants (from reference setup_inputs)
#define B_SZ   512
#define L_CL   20
#define D_F    128
#define N_TOT  (B_SZ * L_CL)          // 10240

// exp(x/T) = exp2(x * log2(e)/T);  ln(S) = log2(S) * ln(2)
__constant__ const float kExpScale = 1.4426950408889634f / 0.07f;
__constant__ const float kInvT     = 1.0f / 0.07f;

typedef unsigned short u16;
typedef __attribute__((ext_vector_type(16))) __bf16 bf16x16;
typedef __attribute__((ext_vector_type(8)))  float  floatx8;

union FragU { bf16x16 v; uint4 q[2]; };

// ---------------------------------------------------------------------------
// Kernel 1: split fp32 features into bf16 hi + bf16 lo (round-to-nearest-even)
// ---------------------------------------------------------------------------
__global__ void supcon_pack_bf16(const float* __restrict__ x,
                                 u16* __restrict__ hi, u16* __restrict__ lo,
                                 int n) {
  int idx = blockIdx.x * blockDim.x + threadIdx.x;
  if (idx >= n) return;
  float f = x[idx];
  uint32_t u  = __float_as_uint(f);
  uint32_t hb = (u + 0x7FFFu + ((u >> 16) & 1u)) >> 16;       // RNE to bf16
  float fh = __uint_as_float(hb << 16);
  float r  = f - fh;                                           // exact residual
  uint32_t u2 = __float_as_uint(r);
  uint32_t lb = (u2 + 0x7FFFu + ((u2 >> 16) & 1u)) >> 16;
  hi[idx] = (u16)hb;
  lo[idx] = (u16)lb;
}

__global__ void supcon_zero(float* __restrict__ p, int n) {
  int idx = blockIdx.x * blockDim.x + threadIdx.x;
  if (idx < n) p[idx] = 0.0f;
}

// ---------------------------------------------------------------------------
// Kernel 2: fused Gram / exp / mask / positive-dot accumulation.
//   grid.x = B/16 (i-tiles), grid.y = 80 (column chunks of 8 col-tiles)
//   block  = 256 threads = 8 wave32; wave w owns col-tile q = 8*blockIdx.y + w
// For its 16 columns (class c = q>>5, j0 = (q&31)*16) each wave:
//   - keeps B-fragments (hi/lo, 4 K-steps) in registers,
//   - loops l = 0..19 row-tiles of rows (l, i0..i0+15),
//   - G-tile = Ah*Bh + Ah*Bl + Al*Bh  (bf16x2 split, fp32 accumulate),
//   - accS += (i!=j) ? exp(G/T) : 0          (-> S[c,i] after lane-reduce)
//   - accP += (l==c && i!=j) ? label[j,c]*G/T : 0   (-> pd[c,i])
// ---------------------------------------------------------------------------
__global__ __launch_bounds__(256)
void supcon_gram(const u16* __restrict__ Fhi, const u16* __restrict__ Flo,
                 const int* __restrict__ labels,
                 float* __restrict__ S_ws, float* __restrict__ pd_ws) {
  const int lane = threadIdx.x & 31;
  const int wave = threadIdx.x >> 5;
  const int half = lane >> 4;          // K-half selector for fragment loads
  const int lr   = lane & 15;          // row (A) / column (B) index in tile

  const int i0 = blockIdx.x * 16;                 // anchor items i0..i0+15
  const int q  = blockIdx.y * 8 + wave;           // col-tile 0..639
  const int c  = q >> 5;                          // class of this column tile
  const int j0 = (q & 31) * 16;                   // first item j in tile

  // ---- load B fragments once (column item j0+lr of class c, all 4 K-steps)
  const long bbase = ((long)(j0 + lr) * L_CL + c) * D_F;
  FragU Bh[4], Bl[4];
#pragma unroll
  for (int ks = 0; ks < 4; ++ks) {
    const u16* ph = Fhi + bbase + ks * 32 + half * 8;
    const u16* pl = Flo + bbase + ks * 32 + half * 8;
    Bh[ks].q[0] = *reinterpret_cast<const uint4*>(ph);
    Bh[ks].q[1] = *reinterpret_cast<const uint4*>(ph + 16);
    Bl[ks].q[0] = *reinterpret_cast<const uint4*>(pl);
    Bl[ks].q[1] = *reinterpret_cast<const uint4*>(pl + 16);
  }
  const float labj = (float)labels[(j0 + lr) * L_CL + c];  // label[j, c]

  float accS[8], accP[8];
#pragma unroll
  for (int v = 0; v < 8; ++v) { accS[v] = 0.0f; accP[v] = 0.0f; }

  for (int l = 0; l < L_CL; ++l) {
    const long abase = ((long)(i0 + lr) * L_CL + l) * D_F;
    if (l + 1 < L_CL) {  // hint next row-tile into cache (global_prefetch_b8)
      __builtin_prefetch(Fhi + abase + D_F, 0, 3);
      __builtin_prefetch(Flo + abase + D_F, 0, 3);
    }
    floatx8 C = {};
#pragma unroll
    for (int ks = 0; ks < 4; ++ks) {
      FragU Ah, Al;
      const u16* ph = Fhi + abase + ks * 32 + half * 8;
      const u16* pl = Flo + abase + ks * 32 + half * 8;
      Ah.q[0] = *reinterpret_cast<const uint4*>(ph);
      Ah.q[1] = *reinterpret_cast<const uint4*>(ph + 16);
      Al.q[0] = *reinterpret_cast<const uint4*>(pl);
      Al.q[1] = *reinterpret_cast<const uint4*>(pl + 16);
      C = __builtin_amdgcn_wmma_f32_16x16x32_bf16(false, Ah.v, false, Bh[ks].v,
                                                  (short)0, C, false, false);
      C = __builtin_amdgcn_wmma_f32_16x16x32_bf16(false, Ah.v, false, Bl[ks].v,
                                                  (short)0, C, false, false);
      C = __builtin_amdgcn_wmma_f32_16x16x32_bf16(false, Al.v, false, Bh[ks].v,
                                                  (short)0, C, false, false);
    }
    const bool isC = (l == c);
#pragma unroll
    for (int v = 0; v < 8; ++v) {
      // C layout: VGPR v holds row (v + 8*half), lane holds column (j0+lr)
      const int irow = i0 + v + 8 * half;
      const int jcol = j0 + lr;
      const bool diag = (irow == jcol);            // exclude i == j (batch idx)
      float g = C[v];
      float e = __builtin_amdgcn_exp2f(g * kExpScale);
      accS[v] += diag ? 0.0f : e;
      if (isC) accP[v] += diag ? 0.0f : (labj * g * kInvT);
    }
  }

  // Reduce over columns: sum across the 16 lanes of each half, then one
  // atomic per (row, target) into the tiny [L, B] accumulators.
#pragma unroll
  for (int v = 0; v < 8; ++v) {
    float s = accS[v];
    float p = accP[v];
#pragma unroll
    for (int m = 1; m < 16; m <<= 1) {
      s += __shfl_xor(s, m, 16);
      p += __shfl_xor(p, m, 16);
    }
    if (lr == 0) {
      const int irow = i0 + v + 8 * half;
      atomicAdd(&S_ws[c * B_SZ + irow], s);
      atomicAdd(&pd_ws[c * B_SZ + irow], p);
    }
  }
}

// ---------------------------------------------------------------------------
// Kernel 3: final scalar reduction.
//   P[c] = sum_j labels[j,c];  per item i:
//   mlpp = sum_{c: lab[i,c]=1} pd[c,i] - (P[c]-1)*ln(S[c,i])
//   avg  = sum_{c: lab[i,c]=1} (P[c]-1)   (0 -> 1)
//   loss = -(T/Tb) * mean_i(mlpp/avg),  T/Tb = 1
// ---------------------------------------------------------------------------
__global__ __launch_bounds__(512)
void supcon_final(const int* __restrict__ labels,
                  const float* __restrict__ S_ws,
                  const float* __restrict__ pd_ws,
                  float* __restrict__ out) {
  __shared__ float Psh[L_CL];
  __shared__ float wsum[16];
  const int tid = threadIdx.x;

  if (tid < L_CL) {
    int cnt = 0;
    for (int j = 0; j < B_SZ; ++j) cnt += labels[j * L_CL + tid];
    Psh[tid] = (float)cnt;
  }
  __syncthreads();

  const int i = tid;                       // 512 threads == B items
  float mlpp = 0.0f, avg = 0.0f;
  for (int ci = 0; ci < L_CL; ++ci) {
    if (labels[i * L_CL + ci]) {
      float cnt = Psh[ci] - 1.0f;
      float lnS = __builtin_amdgcn_logf(S_ws[ci * B_SZ + i]) * 0.6931471805599453f;
      mlpp += pd_ws[ci * B_SZ + i] - cnt * lnS;
      avg  += cnt;
    }
  }
  float contrib = mlpp / (avg == 0.0f ? 1.0f : avg);

  for (int m = 1; m < 32; m <<= 1) contrib += __shfl_xor(contrib, m, 32);
  if ((tid & 31) == 0) wsum[tid >> 5] = contrib;
  __syncthreads();
  if (tid < 16) {
    float t = wsum[tid];
    for (int m = 1; m < 16; m <<= 1) t += __shfl_xor(t, m, 16);
    if (tid == 0) out[0] = -t / (float)B_SZ;   // TEMP/BASE_TEMP == 1
  }
}

// ---------------------------------------------------------------------------
extern "C" void kernel_launch(void* const* d_in, const int* in_sizes, int n_in,
                              void* d_out, int out_size, void* d_ws, size_t ws_size,
                              hipStream_t stream) {
  const float* features = (const float*)d_in[0];   // [B, L, D] f32
  const int*   labels   = (const int*)d_in[1];     // [B, L] i32
  float* out = (float*)d_out;

  // Workspace layout (~5.33 MB total)
  char* ws = (char*)d_ws;
  u16* Fhi = (u16*)ws;                                       // N*D bf16 hi
  u16* Flo = Fhi + (size_t)N_TOT * D_F;                      // N*D bf16 lo
  float* S_ws  = (float*)(Flo + (size_t)N_TOT * D_F);        // [L, B]
  float* pd_ws = S_ws + (size_t)L_CL * B_SZ;                 // [L, B]

  const int nel = N_TOT * D_F;
  supcon_pack_bf16<<<(nel + 255) / 256, 256, 0, stream>>>(features, Fhi, Flo, nel);
  supcon_zero<<<(2 * L_CL * B_SZ + 255) / 256, 256, 0, stream>>>(S_ws, 2 * L_CL * B_SZ);

  dim3 grid(B_SZ / 16, (N_TOT / 16) / 8);   // 32 x 80 workgroups
  supcon_gram<<<grid, 256, 0, stream>>>(Fhi, Flo, labels, S_ws, pd_ws);

  supcon_final<<<1, 512, 0, stream>>>(labels, S_ws, pd_ws, out);
}
